// Controlled_VQVAE_34608846471823
// MI455X (gfx1250) — matrix operand order repaired and checked by reading, not compiled, not run on hardware
//
#include <hip/hip_runtime.h>
#include <hip/hip_bf16.h>
#include <math.h>

typedef __bf16 bf16_t;
typedef __attribute__((ext_vector_type(16))) __bf16 v16bf;
typedef __attribute__((ext_vector_type(8)))  __bf16 v8bf;
typedef __attribute__((ext_vector_type(8)))  float  v8f;

#define D_DIM   1024
#define K_CODES 8192
#define N_ROWS  (64 * 512)          // 32768
#define M_TILE  32                  // rows per block (2 WMMA M-tiles)
#define LDSA_STRIDE 1032            // 1024 + 8 bf16 pad -> conflict-free ds_load_b128

// ---------------------------------------------------------------------------
// Kernel 1: embedding fp32 -> bf16, plus per-code squared norm
// ---------------------------------------------------------------------------
__global__ __launch_bounds__(256) void conv_embed_kernel(
    const float* __restrict__ e, bf16_t* __restrict__ eb, float* __restrict__ enorm) {
  const int k = blockIdx.x;
  const float* row = e + (size_t)k * D_DIM;
  bf16_t* out = eb + (size_t)k * D_DIM;
  float ss = 0.f;
  for (int j = threadIdx.x; j < D_DIM; j += 256) {
    float v = row[j];
    out[j] = (bf16_t)v;
    ss += v * v;
  }
  #pragma unroll
  for (int off = 16; off >= 1; off >>= 1) ss += __shfl_xor(ss, off, 32);
  __shared__ float sred[8];
  if ((threadIdx.x & 31) == 0) sred[threadIdx.x >> 5] = ss;
  __syncthreads();
  if (threadIdx.x == 0) {
    float t = 0.f;
    #pragma unroll
    for (int w = 0; w < 8; ++w) t += sred[w];
    enorm[k] = t;
  }
}

// ---------------------------------------------------------------------------
// Kernel 2: zero the histogram (ws is poisoned; re-zero every call)
// ---------------------------------------------------------------------------
__global__ void zero_kernel(int* __restrict__ counts) {
  int i = blockIdx.x * blockDim.x + threadIdx.x;
  if (i < K_CODES) counts[i] = 0;
}

// ---------------------------------------------------------------------------
// Kernel 3: WMMA bf16 distance GEMM + fused row-wise argmin
//   block = 32 rows of x, 8 waves; each wave computes a 32x64 C tile:
//   2 A fragments x 4 B fragments -> 8 WMMAs per k-step, B reused across M.
//   dist(m,n) = ||e_n||^2 - 2 * x_m . e_n   (||x||^2 constant per row)
// ---------------------------------------------------------------------------
__global__ __launch_bounds__(256) void vq_argmin_kernel(
    const float*  __restrict__ x,      // [N, D] fp32
    const bf16_t* __restrict__ eb,     // [K, D] bf16 row-major
    const float*  __restrict__ enorm,  // [K]
    int*          __restrict__ out_idx) {
  extern __shared__ char smem[];
  bf16_t* sA   = (bf16_t*)smem;                                    // 32*1032 bf16
  float*  sMin = (float*)(smem + (size_t)M_TILE * LDSA_STRIDE * 2);// [8][32]
  int*    sIdx = (int*)((char*)sMin + 8 * M_TILE * 4);             // [8][32]

  const int tid  = threadIdx.x;
  const int lane = tid & 31;
  const int wave = tid >> 5;
  const int row0 = blockIdx.x * M_TILE;

  // Stage 32 rows of x into LDS as bf16 (padded stride).
  for (int i = tid; i < M_TILE * D_DIM; i += 256) {
    const int r = i >> 10, c = i & (D_DIM - 1);
    sA[r * LDSA_STRIDE + c] = (bf16_t)x[(size_t)(row0 + r) * D_DIM + c];
  }
  __syncthreads();

  const int n16 = lane & 15;       // column-within-tile / row-of-A for this lane
  const int hi  = lane >> 4;       // 0: lanes 0-15, 1: lanes 16-31
  // A fragment source (contiguous K order, matching B):
  //   lanes<16: K = [kb..kb+15];  lanes>=16: K = [kb+16..kb+31]
  const bf16_t* aRow0 = sA + n16 * LDSA_STRIDE + (hi ? 16 : 0);
  const bf16_t* aRow1 = aRow0 + 16 * LDSA_STRIDE;

  float minv[2][8];
  int   mini[2][8];
  #pragma unroll
  for (int mt = 0; mt < 2; ++mt)
    #pragma unroll
    for (int r = 0; r < 8; ++r) { minv[mt][r] = 3.4e38f; mini[mt][r] = 0; }

  for (int t = 0; t < K_CODES / 512; ++t) {
    const int cb = wave * 64 + 512 * t;      // this wave's 64-code group
    v8f acc[2][4];
    #pragma unroll
    for (int mt = 0; mt < 2; ++mt)
      #pragma unroll
      for (int j = 0; j < 4; ++j)
        acc[mt][j] = (v8f){0.f,0.f,0.f,0.f,0.f,0.f,0.f,0.f};

    // B fragment source (contiguous k order, same as A): lane n16 = column (code),
    // lanes 0-15 hold K=kb..kb+15, lanes 16-31 hold K=kb+16..kb+31 (contig 32B).
    const bf16_t* bBase = eb + (size_t)(cb + n16) * D_DIM + hi * 16;
    if (t + 1 < K_CODES / 512)
      __builtin_prefetch(bBase + (size_t)512 * D_DIM, 0, 0);

    for (int kb = 0; kb < D_DIM; kb += 32) {
      v8bf a0lo = *(const v8bf*)(aRow0 + kb);
      v8bf a0hi = *(const v8bf*)(aRow0 + kb + 8);
      v8bf a1lo = *(const v8bf*)(aRow1 + kb);
      v8bf a1hi = *(const v8bf*)(aRow1 + kb + 8);
      v16bf a0 = __builtin_shufflevector(a0lo, a0hi,
                   0,1,2,3,4,5,6,7,8,9,10,11,12,13,14,15);
      v16bf a1 = __builtin_shufflevector(a1lo, a1hi,
                   0,1,2,3,4,5,6,7,8,9,10,11,12,13,14,15);
      #pragma unroll
      for (int j = 0; j < 4; ++j) {
        v16bf b = *(const v16bf*)(bBase + (size_t)j * 16 * D_DIM + kb);
        acc[0][j] = __builtin_amdgcn_wmma_f32_16x16x32_bf16(
            false, a0, false, b, (short)0, acc[0][j], false, false);
        acc[1][j] = __builtin_amdgcn_wmma_f32_16x16x32_bf16(
            false, a1, false, b, (short)0, acc[1][j], false, false);
      }
    }

    // C layout: VGPR r, lanes 0-15 -> (M=r, N=lane); lanes 16-31 -> (M=8+r).
    #pragma unroll
    for (int j = 0; j < 4; ++j) {
      const int col = cb + j * 16 + n16;
      const float en = enorm[col];
      #pragma unroll
      for (int mt = 0; mt < 2; ++mt) {
        #pragma unroll
        for (int r = 0; r < 8; ++r) {
          const float dist = en - 2.0f * acc[mt][j][r];
          if (dist < minv[mt][r]) { minv[mt][r] = dist; mini[mt][r] = col; }
        }
      }
    }
  }

  // Reduce across the 16 lanes that share a row (xor 8,4,2,1 stays in halves).
  #pragma unroll
  for (int mt = 0; mt < 2; ++mt) {
    #pragma unroll
    for (int r = 0; r < 8; ++r) {
      #pragma unroll
      for (int off = 8; off >= 1; off >>= 1) {
        const float ov = __shfl_xor(minv[mt][r], off, 32);
        const int   oi = __shfl_xor(mini[mt][r], off, 32);
        if (ov < minv[mt][r] || (ov == minv[mt][r] && oi < mini[mt][r])) {
          minv[mt][r] = ov; mini[mt][r] = oi;
        }
      }
    }
  }
  if (n16 == 0) {   // lane 0: rows mt*16 + 0..7 ; lane 16: rows mt*16 + 8..15
    #pragma unroll
    for (int mt = 0; mt < 2; ++mt) {
      #pragma unroll
      for (int r = 0; r < 8; ++r) {
        sMin[wave * M_TILE + mt * 16 + hi * 8 + r] = minv[mt][r];
        sIdx[wave * M_TILE + mt * 16 + hi * 8 + r] = mini[mt][r];
      }
    }
  }
  __syncthreads();
  if (tid < M_TILE) {
    float bv = sMin[tid]; int bi = sIdx[tid];
    #pragma unroll
    for (int w = 1; w < 8; ++w) {
      const float v = sMin[w * M_TILE + tid]; const int i2 = sIdx[w * M_TILE + tid];
      if (v < bv || (v == bv && i2 < bi)) { bv = v; bi = i2; }
    }
    out_idx[row0 + tid] = bi;
  }
}

// ---------------------------------------------------------------------------
// Kernel 4: gather quantized rows (fp32 codebook), per-row MSE partial,
// histogram, and indices-as-float output. One block per row.
// ---------------------------------------------------------------------------
__global__ __launch_bounds__(256) void gather_kernel(
    const float* __restrict__ x, const float* __restrict__ e,
    const int* __restrict__ idx, float* __restrict__ outq,
    float* __restrict__ out_idxf, int* __restrict__ counts,
    float* __restrict__ partial) {
  const int row = blockIdx.x;
  const int k = idx[row];
  const float* er = e + (size_t)k * D_DIM;
  const float* xr = x + (size_t)row * D_DIM;
  float* qr = outq + (size_t)row * D_DIM;
  float ss = 0.f;
  for (int j = threadIdx.x; j < D_DIM; j += 256) {
    const float q = er[j];
    qr[j] = q;                          // straight-through output == quantized
    const float d = q - xr[j];
    ss += d * d;
  }
  #pragma unroll
  for (int off = 16; off >= 1; off >>= 1) ss += __shfl_xor(ss, off, 32);
  __shared__ float sred[8];
  if ((threadIdx.x & 31) == 0) sred[threadIdx.x >> 5] = ss;
  __syncthreads();
  if (threadIdx.x == 0) {
    float t = 0.f;
    #pragma unroll
    for (int w = 0; w < 8; ++w) t += sred[w];
    partial[row] = t;                   // deterministic per-row partial
    atomicAdd(&counts[k], 1);           // integer atomic: deterministic
    out_idxf[row] = (float)k;
  }
}

// ---------------------------------------------------------------------------
// Kernel 5: finalize loss and perplexity (single block, deterministic)
// ---------------------------------------------------------------------------
__global__ __launch_bounds__(256) void finalize_kernel(
    const int* __restrict__ counts, const float* __restrict__ partial,
    float* __restrict__ loss_out, float* __restrict__ ppl_out) {
  __shared__ float sA[256], sB[256];
  float mse = 0.f, ent = 0.f;
  const float invN = 1.0f / (float)N_ROWS;
  for (int i = threadIdx.x; i < N_ROWS; i += 256) mse += partial[i];
  for (int k = threadIdx.x; k < K_CODES; k += 256) {
    const float p = (float)counts[k] * invN;
    ent += p * logf(p + 1e-10f);
  }
  sA[threadIdx.x] = mse; sB[threadIdx.x] = ent;
  __syncthreads();
  for (int off = 128; off >= 1; off >>= 1) {
    if (threadIdx.x < off) { sA[threadIdx.x] += sA[threadIdx.x + off];
                             sB[threadIdx.x] += sB[threadIdx.x + off]; }
    __syncthreads();
  }
  if (threadIdx.x == 0) {
    // q_latent + 0.25*e_latent, both numerically mean((q-x)^2)
    *loss_out = 1.25f * sA[0] / (float)((size_t)N_ROWS * D_DIM);
    *ppl_out  = expf(-sB[0]);
  }
}

// ---------------------------------------------------------------------------
extern "C" void kernel_launch(void* const* d_in, const int* in_sizes, int n_in,
                              void* d_out, int out_size, void* d_ws, size_t ws_size,
                              hipStream_t stream) {
  const float* x = (const float*)d_in[0];   // [B,T,D] fp32
  const float* e = (const float*)d_in[1];   // [K,D]   fp32

  char* ws = (char*)d_ws;
  bf16_t* eb    = (bf16_t*)ws;                                   // 16 MB
  float*  enorm = (float*)(ws + (size_t)K_CODES * D_DIM * 2);    // 32 KB
  int*    idx   = (int*)  ((char*)enorm + (size_t)K_CODES * 4);  // 128 KB
  int*    counts= (int*)  ((char*)idx   + (size_t)N_ROWS * 4);   // 32 KB
  float*  part  = (float*)((char*)counts + (size_t)K_CODES * 4); // 128 KB

  float* out = (float*)d_out;
  const size_t ND = (size_t)N_ROWS * D_DIM;
  float* loss_out = out;                 // [0]
  float* outq     = out + 1;             // [1 .. 1+ND)
  float* ppl_out  = out + 1 + ND;        // [1+ND]
  float* idxf     = out + 2 + ND;        // [2+ND .. 2+ND+N)

  // Dynamic LDS: A-stage (32 x 1032 bf16 = 66048 B) + cross-wave argmin arrays.
  const size_t shmem = (size_t)M_TILE * LDSA_STRIDE * 2   // sA
                     + 8 * M_TILE * 4                     // sMin
                     + 8 * M_TILE * 4;                    // sIdx

  zero_kernel<<<(K_CODES + 255) / 256, 256, 0, stream>>>(counts);
  conv_embed_kernel<<<K_CODES, 256, 0, stream>>>(e, eb, enorm);
  vq_argmin_kernel<<<N_ROWS / M_TILE, 256, shmem, stream>>>(x, eb, enorm, idx);
  gather_kernel<<<N_ROWS, 256, 0, stream>>>(x, e, idx, outq, idxf, counts, part);
  finalize_kernel<<<1, 256, 0, stream>>>(counts, part, loss_out, ppl_out);
}
